// MusicTransformer_5531917877524
// MI455X (gfx1250) — compile-verified, matrix operand order checked by reading, not run on hardware
//
#include <hip/hip_runtime.h>
#include <stdint.h>

#define LSEQ    2048
#define DMODEL  256
#define NVOCAB  240
#define NLAYERS 6
#define NBATCH  4
#define FC_K     (LSEQ*DMODEL)     // 524288
#define FC_CHUNK 4096
#define FC_NCHUNK (FC_K/FC_CHUNK)  // 128

typedef __attribute__((ext_vector_type(16))) __bf16       v16bf;
typedef __attribute__((ext_vector_type(8)))  float        v8f;
typedef __attribute__((ext_vector_type(4)))  unsigned int u32x4;

static __device__ __forceinline__ uint16_t f2bf(float f) {
  uint32_t u = __float_as_uint(f);
  uint32_t r = (u + 0x7FFFu + ((u >> 16) & 1u)) >> 16;   // round-to-nearest-even
  return (uint16_t)r;
}
static __device__ __forceinline__ float bf2f(uint16_t h) {
  return __uint_as_float(((uint32_t)h) << 16);
}

// ---------------- weights: fp32 [K,N] -> bf16 transposed [N,K], per layer ----------------
__global__ void k_w_to_bf16_t(const float* __restrict__ in, uint16_t* __restrict__ out) {
  int o = blockIdx.x * blockDim.x + threadIdx.x;   // over NLAYERS*DMODEL*DMODEL, out-linear
  int k = o & (DMODEL - 1);
  int n = (o >> 8) & (DMODEL - 1);
  int layer = o >> 16;
  out[o] = f2bf(in[(size_t)layer * DMODEL * DMODEL + (size_t)k * DMODEL + n]);
}

// ---------------- bf16 [L,D] -> bf16 [D,L] (LDS 32x32 tile transpose) ----------------
__global__ void k_transpose_bf16(const uint16_t* __restrict__ in, uint16_t* __restrict__ out) {
  __shared__ uint16_t tile[32][33];
  int b  = blockIdx.z;
  int r0 = blockIdx.x * 32;      // L
  int c0 = blockIdx.y * 32;      // D
  const uint16_t* ib = in  + (size_t)b * LSEQ * DMODEL;
  uint16_t*       ob = out + (size_t)b * LSEQ * DMODEL;
  int tx = threadIdx.x, ty = threadIdx.y;          // 32 x 8
#pragma unroll
  for (int j = 0; j < 32; j += 8)
    tile[ty + j][tx] = ib[(size_t)(r0 + ty + j) * DMODEL + c0 + tx];
  __syncthreads();
#pragma unroll
  for (int j = 0; j < 32; j += 8)
    ob[(size_t)(c0 + ty + j) * LSEQ + r0 + tx] = tile[tx][ty + j];
}

// ---------------- embedding + sinusoidal positional encoding ----------------
__global__ void k_embed(const int* __restrict__ tokens, const float* __restrict__ table,
                        float* __restrict__ xf, uint16_t* __restrict__ xb) {
  int i  = blockIdx.x * blockDim.x + threadIdx.x;     // over NBATCH*LSEQ*DMODEL
  int d  = i & (DMODEL - 1);
  int bl = i >> 8;                                    // b*LSEQ + l
  int l  = bl & (LSEQ - 1);
  int tok = tokens[bl];
  float ang = powf((float)l * 1e-5f, (float)d * (1.0f / DMODEL));
  if ((d + 1) & 1) ang -= 1.5707963267948966f;        // even d: subtract pi/2
  float v = table[tok * DMODEL + d] + sinf(ang);
  xf[i] = v;
  xb[i] = f2bf(v);
}

// ---------------- q2[r] = ||Q[r,:]||^2 (one wave per row) ----------------
__global__ void k_rownorm2(const uint16_t* __restrict__ Q, float* __restrict__ q2) {
  int lane = threadIdx.x & 31, wid = threadIdx.x >> 5;
  int row = blockIdx.x * 8 + wid;
  union { u32x4 q; uint16_t s[8]; } u;
  u.q = *(const u32x4*)(Q + (size_t)row * DMODEL + lane * 8);
  float acc = 0.f;
#pragma unroll
  for (int j = 0; j < 8; ++j) { float f = bf2f(u.s[j]); acc += f * f; }
#pragma unroll
  for (int o = 16; o > 0; o >>= 1) acc += __shfl_xor(acc, o, 32);
  if (lane == 0) q2[row] = acc;
}

// ---------------- WMMA bf16 GEMM, 64x32 tile per wave, B always NxK ----------------
static __device__ __forceinline__ v16bf ldfrag(const uint16_t* p) {
  union { v16bf v; u32x4 q[2]; } u;
  u.q[0] = *(const u32x4*)(p);        // K = kh .. kh+7
  u.q[1] = *(const u32x4*)(p + 16);   // K = kh+16 .. kh+23
  return u.v;
}
static __device__ __forceinline__ v8f wmma_bf16(v16bf a, v16bf b, v8f c) {
  return __builtin_amdgcn_wmma_f32_16x16x32_bf16(false, a, false, b, (short)0, c, false, false);
}

// D = A(MxK) * Bt(NxK)^T [+bias][leaky 0.2][+residual]; writes fp32 and/or bf16.
__global__ void __launch_bounds__(256)
k_gemm_wmma(const uint16_t* __restrict__ A, int lda, size_t strideA,
            const uint16_t* __restrict__ Bt, int ldb, size_t strideB,
            float* outF, uint16_t* outB, const float* residual, size_t strideO,
            const float* bias, int M, int N, int K, int act) {
  int lane = threadIdx.x & 31, wid = threadIdx.x >> 5;
  int tiles_n = N >> 5;                                // 32-wide col tiles
  int t = blockIdx.x * 8 + wid;
  if (t >= (M >> 6) * tiles_n) return;                 // wave-uniform; 64-tall row tiles
  int tm = t / tiles_n, tn = t % tiles_n;
  int r0 = tm << 6, c0 = tn << 5;
  const uint16_t* Ab = A  + (size_t)blockIdx.z * strideA;
  const uint16_t* Bb = Bt + (size_t)blockIdx.z * strideB;
  int l16 = lane & 15;
  int kh  = (lane >> 4) * 8;
  const uint16_t* arow[4];
#pragma unroll
  for (int mi = 0; mi < 4; ++mi)
    arow[mi] = Ab + (size_t)(r0 + 16 * mi + l16) * lda + kh;
  const uint16_t* brow0 = Bb + (size_t)(c0 + l16) * ldb + kh;
  const uint16_t* brow1 = brow0 + (size_t)16 * ldb;
  v8f acc[4][2] = {};
  for (int k0 = 0; k0 < K; k0 += 32) {
    v16bf b0 = ldfrag(brow0 + k0), b1 = ldfrag(brow1 + k0);
#pragma unroll
    for (int mi = 0; mi < 4; ++mi) {
      v16bf a = ldfrag(arow[mi] + k0);
      acc[mi][0] = wmma_bf16(a, b0, acc[mi][0]);
      acc[mi][1] = wmma_bf16(a, b1, acc[mi][1]);
    }
  }
  float*    oF = outF ? outF + (size_t)blockIdx.z * strideO : nullptr;
  uint16_t* oB = outB ? outB + (size_t)blockIdx.z * strideO : nullptr;
  const float* rs = residual ? residual + (size_t)blockIdx.z * strideO : nullptr;
  int mhi = (lane >> 4) * 8;
#pragma unroll
  for (int mi = 0; mi < 4; ++mi)
#pragma unroll
    for (int ni = 0; ni < 2; ++ni) {
      int col = c0 + ni * 16 + l16;
      float bv = bias ? bias[col] : 0.f;
#pragma unroll
      for (int e = 0; e < 8; ++e) {
        int row = r0 + mi * 16 + mhi + e;
        float v = acc[mi][ni][e] + bv;
        if (act) v = (v > 0.f) ? v : 0.2f * v;       // leaky relu 0.2
        size_t idx = (size_t)row * N + col;
        if (rs) v += rs[idx];
        if (oF) oF[idx] = v;
        if (oB) oB[idx] = f2bf(v);
      }
    }
}

// ---------------- fused skew + Srel + scale + softmax -> bf16 probs ----------------
// Srel[i,c] = (c<=i)   ? S[i, L-1-i+c] - q2[i]
//           : (c==i+1) ? 0
//           :            S[i+1, c-i-2] - q2[i+1]
__global__ void __launch_bounds__(256)
k_softmax_rel(const float* __restrict__ S, const float* __restrict__ q2,
              uint16_t* __restrict__ P) {
  int i = blockIdx.x, b = blockIdx.y, t = threadIdx.x;
  const float* Si  = S + ((size_t)b * LSEQ + i) * LSEQ;
  const float* Si1 = Si + LSEQ;
  float q2i = q2[b * LSEQ + i];
  float q2n = (i + 1 < LSEQ) ? q2[b * LSEQ + i + 1] : 0.f;
  float v[8];
  float mx = -3.4e38f;
#pragma unroll
  for (int j = 0; j < 8; ++j) {
    int c = t + j * 256;
    float srel;
    if (c <= i)          srel = Si[LSEQ - 1 - i + c] - q2i;
    else if (c == i + 1) srel = 0.f;
    else                 srel = Si1[c - i - 2] - q2n;
    float s = (Si[c] + srel) * 0.125f;               // / sqrt(64)
    v[j] = s;
    mx = fmaxf(mx, s);
  }
  __shared__ float red[256];
  red[t] = mx; __syncthreads();
  for (int s1 = 128; s1 > 0; s1 >>= 1) { if (t < s1) red[t] = fmaxf(red[t], red[t + s1]); __syncthreads(); }
  mx = red[0]; __syncthreads();
  float sum = 0.f;
#pragma unroll
  for (int j = 0; j < 8; ++j) { v[j] = __expf(v[j] - mx); sum += v[j]; }
  red[t] = sum; __syncthreads();
  for (int s1 = 128; s1 > 0; s1 >>= 1) { if (t < s1) red[t] += red[t + s1]; __syncthreads(); }
  float inv = 1.f / red[0];
  uint16_t* Pr = P + ((size_t)b * LSEQ + i) * LSEQ;
#pragma unroll
  for (int j = 0; j < 8; ++j) Pr[t + j * 256] = f2bf(v[j] * inv);
}

// ---------------- final FC: deterministic two-stage fp32 dot ----------------
__global__ void k_fc_partial(const float* __restrict__ x, const float* __restrict__ Wfc,
                             float* __restrict__ partial) {
  int chunk = blockIdx.x, b = blockIdx.y, n = threadIdx.x;
  if (n >= NVOCAB) return;
  const float* xb = x + (size_t)b * FC_K;
  int k0 = chunk * FC_CHUNK;
  float acc = 0.f;
  for (int k = k0; k < k0 + FC_CHUNK; ++k)
    acc = fmaf(xb[k], Wfc[(size_t)k * NVOCAB + n], acc);
  partial[((size_t)b * FC_NCHUNK + chunk) * NVOCAB + n] = acc;
}

__global__ void k_fc_softmax(const float* __restrict__ partial, const float* __restrict__ bfc,
                             float* __restrict__ out) {
  int b = blockIdx.x, t = threadIdx.x;
  float val = 0.f, logit = -3.4e38f;
  if (t < NVOCAB) {
    val = bfc[t];
    for (int c = 0; c < FC_NCHUNK; ++c)
      val += partial[((size_t)b * FC_NCHUNK + c) * NVOCAB + t];
    logit = val;
  }
  __shared__ float red[256];
  red[t] = logit; __syncthreads();
  for (int s = 128; s > 0; s >>= 1) { if (t < s) red[t] = fmaxf(red[t], red[t + s]); __syncthreads(); }
  float mx = red[0]; __syncthreads();
  float e = (t < NVOCAB) ? __expf(val - mx) : 0.f;
  red[t] = e; __syncthreads();
  for (int s = 128; s > 0; s >>= 1) { if (t < s) red[t] += red[t + s]; __syncthreads(); }
  if (t < NVOCAB) out[b * NVOCAB + t] = e / red[0];
}

// ---------------- host orchestration ----------------
extern "C" void kernel_launch(void* const* d_in, const int* in_sizes, int n_in,
                              void* d_out, int out_size, void* d_ws, size_t ws_size,
                              hipStream_t stream) {
  (void)in_sizes; (void)n_in; (void)out_size; (void)ws_size;
  const int*   tokens = (const int*)d_in[0];
  const float* table  = (const float*)d_in[1];
  const float* w_src[8] = { (const float*)d_in[2], (const float*)d_in[3], (const float*)d_in[4],
                            (const float*)d_in[5], (const float*)d_in[6], (const float*)d_in[7],
                            (const float*)d_in[8], (const float*)d_in[10] };
  const float* bf1 = (const float*)d_in[9];
  const float* bf2 = (const float*)d_in[11];
  const float* Wfc = (const float*)d_in[12];
  const float* bfc = (const float*)d_in[13];
  float* out = (float*)d_out;

  uint8_t* ws = (uint8_t*)d_ws;
  size_t off = 0;
  auto carve = [&](size_t bytes) -> void* {
    void* p = ws + off; off += (bytes + 255) & ~(size_t)255; return p;
  };
  const size_t BLD = (size_t)NBATCH * LSEQ * DMODEL;
  const size_t LD  = (size_t)LSEQ * DMODEL;
  const size_t LL  = (size_t)LSEQ * LSEQ;
  const size_t WSZ = (size_t)NLAYERS * DMODEL * DMODEL;

  float*    x_f  = (float*)carve(BLD * 4);
  uint16_t* x_b  = (uint16_t*)carve(BLD * 2);
  uint16_t* q_b  = (uint16_t*)carve(BLD * 2);
  uint16_t* k_b  = (uint16_t*)carve(BLD * 2);
  uint16_t* v_b  = (uint16_t*)carve(BLD * 2);
  uint16_t* vt_b = (uint16_t*)carve(BLD * 2);   // V transposed [D,L] per batch
  uint16_t* h_b  = (uint16_t*)carve(BLD * 2);
  float*    q2   = (float*)carve((size_t)NBATCH * LSEQ * 4);
  float*    S    = (float*)carve((size_t)NBATCH * LL * 4);
  uint16_t* P    = (uint16_t*)carve((size_t)NBATCH * LL * 2);
  uint16_t* w_b[8];                              // all weights bf16, transposed [N,K]
  for (int i = 0; i < 8; ++i) w_b[i] = (uint16_t*)carve(WSZ * 2);
  float* partial = (float*)carve((size_t)NBATCH * FC_NCHUNK * NVOCAB * 4);

  // weights -> bf16 + transpose, once per call (deterministic)
  for (int i = 0; i < 8; ++i)
    k_w_to_bf16_t<<<(unsigned)((WSZ + 255) / 256), 256, 0, stream>>>(w_src[i], w_b[i]);

  k_embed<<<(unsigned)((BLD + 255) / 256), 256, 0, stream>>>(tokens, table, x_f, x_b);

  // generic launcher: Bt supplied as [N,K]
  auto gemm = [&](const uint16_t* A, int lda, size_t sA, const uint16_t* Bt, int ldb, size_t sB,
                  float* oF, uint16_t* oB, const float* res, size_t sO, const float* bias,
                  int M, int N, int K, int act) {
    dim3 g((unsigned)(((M >> 6) * (N >> 5)) / 8), 1, NBATCH);
    k_gemm_wmma<<<g, 256, 0, stream>>>(A, lda, sA, Bt, ldb, sB, oF, oB, res, sO, bias, M, N, K, act);
  };

  for (int layer = 0; layer < NLAYERS; ++layer) {
    for (int half = 0; half < 2; ++half) {
      const uint16_t* wq = w_b[half ? 3 : 0] + (size_t)layer * DMODEL * DMODEL;
      const uint16_t* wk = w_b[half ? 4 : 1] + (size_t)layer * DMODEL * DMODEL;
      const uint16_t* wv = w_b[half ? 5 : 2] + (size_t)layer * DMODEL * DMODEL;
      // Q/K/V projections: [L,D] = [L,D] x Wt[D,D]^T
      gemm(x_b, DMODEL, LD, wq, DMODEL, 0, nullptr, q_b, nullptr, LD, nullptr, LSEQ, DMODEL, DMODEL, 0);
      gemm(x_b, DMODEL, LD, wk, DMODEL, 0, nullptr, k_b, nullptr, LD, nullptr, LSEQ, DMODEL, DMODEL, 0);
      gemm(x_b, DMODEL, LD, wv, DMODEL, 0, nullptr, v_b, nullptr, LD, nullptr, LSEQ, DMODEL, DMODEL, 0);
      k_rownorm2<<<(unsigned)(NBATCH * LSEQ / 8), 256, 0, stream>>>(q_b, q2);
      // Vt[D,L] for the contiguous-B PV GEMM
      k_transpose_bf16<<<dim3(LSEQ / 32, DMODEL / 32, NBATCH), dim3(32, 8), 0, stream>>>(v_b, vt_b);
      // S = Q @ K^T : [L,L]  (K is naturally the [N,K] operand)
      gemm(q_b, DMODEL, LD, k_b, DMODEL, LD, S, nullptr, nullptr, LL, nullptr, LSEQ, LSEQ, DMODEL, 0);
      // fused skew/Srel + softmax -> bf16 probs
      k_softmax_rel<<<dim3(LSEQ, NBATCH), 256, 0, stream>>>(S, q2, P);
      // x = P @ V + x (in-place residual; writes fp32 + bf16 views of x)
      gemm(P, LSEQ, LL, vt_b, LSEQ, LD, x_f, x_b, x_f, LD, nullptr, LSEQ, DMODEL, LSEQ, 0);
    }
    const uint16_t* wf1 = w_b[6] + (size_t)layer * DMODEL * DMODEL;
    const uint16_t* wf2 = w_b[7] + (size_t)layer * DMODEL * DMODEL;
    // h = leaky_relu(x@Wf1 + bf1)
    gemm(x_b, DMODEL, LD, wf1, DMODEL, 0, nullptr, h_b, nullptr, LD, bf1 + layer * DMODEL,
         LSEQ, DMODEL, DMODEL, 1);
    // x = h@Wf2 + bf2 (no residual)
    gemm(h_b, DMODEL, LD, wf2, DMODEL, 0, x_f, x_b, nullptr, LD, bf2 + layer * DMODEL,
         LSEQ, DMODEL, DMODEL, 0);
  }

  k_fc_partial<<<dim3(FC_NCHUNK, NBATCH), 256, 0, stream>>>(x_f, Wfc, partial);
  k_fc_softmax<<<NBATCH, 256, 0, stream>>>(partial, bfc, out);
}
